// MultiLatentAttention_34840774705272
// MI455X (gfx1250) — matrix operand order repaired and checked by reading, not compiled, not run on hardware
//
#include <hip/hip_runtime.h>
#include <hip/hip_bf16.h>
#include <cstdint>
#include <cstddef>

typedef __attribute__((ext_vector_type(16))) __bf16 bf16x16;
typedef __attribute__((ext_vector_type(8)))  float  f32x8;

#define B_SZ   2
#define S_SZ   2048
#define H_SZ   16
#define DIN    2048
#define DKV    512
#define DHEAD  128
#define MROWS  (B_SZ * S_SZ)   /* 4096 token rows */

static __device__ __forceinline__ f32x8 wmma_bf16(bf16x16 a, bf16x16 b, f32x8 c) {
  return __builtin_amdgcn_wmma_f32_16x16x32_bf16(false, a, false, b, (short)0, c, false, false);
}

// ---------------------------------------------------------------------------
// fp32 -> bf16 conversion
// ---------------------------------------------------------------------------
__global__ void f32_to_bf16_kernel(const float* __restrict__ src,
                                   __bf16* __restrict__ dst, int n) {
  int i = blockIdx.x * blockDim.x + threadIdx.x;
  if (i < n) dst[i] = (__bf16)src[i];
}

// ---------------------------------------------------------------------------
// Generic bf16 WMMA GEMM: C[M=4096,N] = A[M,K] @ B[K,N] + bias
// Block tile 64(M) x 128(N), K-step 32, 256 threads = 8 waves in 2x4;
// each wave owns a 32x32 tile -> 4 WMMAs per k-step.
// B is staged TRANSPOSED into LDS (BsT[n][k]) via dword pair-packing so that
// B-fragment reads are contiguous 32B (merge to 2x ds_load_b128).
// ALL shape/output parameters are template constants so the epilogue folds to
// immediate-offset stores (no runtime division, no pointer-null branches).
// ---------------------------------------------------------------------------
template <int N, int K, bool F32OUT, bool BF16OUT,
          int OSTRIDE, int G_IN, int G_OUT, int G_OFF>
__global__ void __launch_bounds__(256)
gemm_bf16_kernel(const __bf16* __restrict__ A, const __bf16* __restrict__ Bm,
                 const float* __restrict__ bias,
                 float* __restrict__ Cf, __bf16* __restrict__ Cb) {
  __shared__ __bf16 As[64][40];     // row-major [m][k], +8 pad
  __shared__ __bf16 BsT[128][40];   // TRANSPOSED [n][k], +8 pad

  const int tid  = threadIdx.x;
  const int wid  = tid >> 5;
  const int lane = tid & 31;
  const int half = lane >> 4;
  const int l15  = lane & 15;
  const int wm   = wid >> 2;        // 0..1 : 32-row group
  const int wn   = wid & 3;         // 0..3 : 32-col group
  const int m0   = blockIdx.y * 64;
  const int n0   = blockIdx.x * 128;

  const f32x8 zero = {0.f, 0.f, 0.f, 0.f, 0.f, 0.f, 0.f, 0.f};
  f32x8 acc[2][2];
#pragma unroll
  for (int i = 0; i < 2; i++)
#pragma unroll
    for (int j = 0; j < 2; j++) acc[i][j] = zero;

  // A staging: 8 contiguous bf16 per thread (one uint4)
  const int ar = (tid * 8) >> 5, ac = (tid * 8) & 31;
  // B staging: thread handles k-pair kp (k=2*kp) x 8 columns (n8 = ng*8)
  const int kp = tid & 15;
  const int ng = tid >> 4;

  for (int k0 = 0; k0 < K; k0 += 32) {
    // ---- stage A (row-major, vectorized) ----
    *reinterpret_cast<uint4*>(&As[ar][ac]) =
        *reinterpret_cast<const uint4*>(A + (size_t)(m0 + ar) * K + (k0 + ac));
    // ---- stage B transposed: read 2 rows x 8 cols, pack (k,k+1) dwords ----
    {
      const __bf16* brow = Bm + (size_t)(k0 + 2 * kp) * N + n0 + ng * 8;
      const uint4 r0 = *reinterpret_cast<const uint4*>(brow);
      const uint4 r1 = *reinterpret_cast<const uint4*>(brow + N);
      const unsigned short* p0 = reinterpret_cast<const unsigned short*>(&r0);
      const unsigned short* p1 = reinterpret_cast<const unsigned short*>(&r1);
#pragma unroll
      for (int j = 0; j < 8; j++) {
        const unsigned int pk = (unsigned int)p0[j] | ((unsigned int)p1[j] << 16);
        *reinterpret_cast<unsigned int*>(&BsT[ng * 8 + j][2 * kp]) = pk;
      }
    }
    __syncthreads();

    bf16x16 af[2], bfv[2];
#pragma unroll
    for (int i = 0; i < 2; i++) {
      const int arow = wm * 32 + i * 16 + l15;
#pragma unroll
      for (int e = 0; e < 16; e++) {
        const int kk = half * 16 + e;
        af[i][e] = As[arow][kk];
      }
    }
#pragma unroll
    for (int j = 0; j < 2; j++) {
      const int brow = wn * 32 + j * 16 + l15;
#pragma unroll
      for (int e = 0; e < 16; e++)
        bfv[j][e] = BsT[brow][half * 16 + e];   // contiguous 32B
    }
#pragma unroll
    for (int i = 0; i < 2; i++)
#pragma unroll
      for (int j = 0; j < 2; j++)
        acc[i][j] = wmma_bf16(af[i], bfv[j], acc[i][j]);
    __syncthreads();
  }

#pragma unroll
  for (int j = 0; j < 2; j++) {
    const int col = n0 + wn * 32 + j * 16 + l15;
    const float bv = bias[col];
    const int oc = (col / G_IN) * G_OUT + G_OFF + (col % G_IN);  // const divisor
#pragma unroll
    for (int i = 0; i < 2; i++) {
      const int row0 = m0 + wm * 32 + i * 16 + half * 8;
      if (F32OUT) {
        float* pf = Cf + (size_t)row0 * N + col;
#pragma unroll
        for (int r = 0; r < 8; r++)
          pf[(size_t)r * N] = acc[i][j][r] + bv;          // immediate offsets
      }
      if (BF16OUT) {
        __bf16* pb = Cb + (size_t)row0 * OSTRIDE + oc;
#pragma unroll
        for (int r = 0; r < 8; r++)
          pb[(size_t)r * OSTRIDE] = (__bf16)(acc[i][j][r] + bv);
      }
    }
  }
}

// ---------------------------------------------------------------------------
// Rotary embedding (fp32 math). One thread per (token-row, head, i<16).
// ---------------------------------------------------------------------------
__global__ void rope_kernel(const float* __restrict__ qpre,
                            const float* __restrict__ kpre,
                            __bf16* __restrict__ qbuf, __bf16* __restrict__ kbuf,
                            float* __restrict__ krot_out) {
  const int idx = blockIdx.x * blockDim.x + threadIdx.x;
  if (idx >= MROWS * H_SZ * 16) return;
  const int i   = idx & 15;
  const int h   = (idx >> 4) & 15;
  const int row = idx >> 8;
  const int s   = row & (S_SZ - 1);

  const float invf = __powf(10000.0f, -(float)i / 16.0f);
  const float ang  = (float)s * invf;
  const float c = __cosf(ang), sn = __sinf(ang);

  const int base = row * (H_SZ * 32) + h * 32;
  const float q1 = qpre[base + i],    q2 = qpre[base + 16 + i];
  const float k1 = kpre[base + i],    k2 = kpre[base + 16 + i];
  const float qo1 = q1 * c - q2 * sn, qo2 = q2 * c + q1 * sn;
  const float ko1 = k1 * c - k2 * sn, ko2 = k2 * c + k1 * sn;

  const size_t qb = (size_t)row * (H_SZ * DHEAD) + h * DHEAD + 96;
  qbuf[qb + i]      = (__bf16)qo1;
  qbuf[qb + 16 + i] = (__bf16)qo2;
  kbuf[qb + i]      = (__bf16)ko1;
  kbuf[qb + 16 + i] = (__bf16)ko2;
  krot_out[base + i]      = ko1;
  krot_out[base + 16 + i] = ko2;
}

// ---------------------------------------------------------------------------
// Flash attention: grid (S/64, H, B), block 128 (4 waves).
// Each wave owns a 16-query tile. 64-key tiles per iteration (32 WMMAs) so
// softmax reductions, O-rescale and barriers amortize over 2x the keys.
// K tile row-major in LDS; V tile staged TRANSPOSED (VtT[d][k]).
// ---------------------------------------------------------------------------
#define KT 64
__global__ void __launch_bounds__(128)
mla_attention_kernel(const __bf16* __restrict__ Q, const __bf16* __restrict__ Kb,
                     const __bf16* __restrict__ Vb, __bf16* __restrict__ O) {
  __shared__ __bf16 Kt[KT][DHEAD + 8];   // [k][d]
  __shared__ __bf16 VtT[DHEAD][KT + 8];  // TRANSPOSED [d][k]
  __shared__ __bf16 Pt[4][16][KT + 8];   // per-wave P staging

  const int b    = blockIdx.z;
  const int h    = blockIdx.y;
  const int wid  = threadIdx.x >> 5;
  const int lane = threadIdx.x & 31;
  const int half = lane >> 4;
  const int l15  = lane & 15;
  const int q0   = blockIdx.x * 64 + wid * 16;

  // Q A-fragments (4 chunks of K=32 along d)
  bf16x16 qf[4];
  const __bf16* qrow = Q + (size_t)(b * S_SZ + q0 + l15) * (H_SZ * DHEAD) + h * DHEAD;
#pragma unroll
  for (int dc = 0; dc < 4; dc++) {
#pragma unroll
    for (int e = 0; e < 16; e++) {
      const int kk = half * 16 + e;
      qf[dc][e] = qrow[dc * 32 + kk];
    }
  }

  const f32x8 zero = {0.f, 0.f, 0.f, 0.f, 0.f, 0.f, 0.f, 0.f};
  f32x8 oacc[8];
#pragma unroll
  for (int d = 0; d < 8; d++) oacc[d] = zero;
  float mrow[8], lrow[8];
#pragma unroll
  for (int r = 0; r < 8; r++) { mrow[r] = -1e30f; lrow[r] = 0.f; }

  const float scale = 0.08838834764831845f;  // 1/sqrt(128)

  // K tile staging: thread -> row kr (0..63), 64-col chunk kc
  const int kr = threadIdx.x >> 1;
  const int kc = (threadIdx.x & 1) * 64;
  // V transposed staging: thread -> k-pair vkp (k=2*vkp, 0..62), 32-d group vdg
  const int vkp = threadIdx.x & 31;
  const int vdg = threadIdx.x >> 5;    // 0..3 -> d0 = vdg*32

  for (int kt0 = 0; kt0 < S_SZ; kt0 += KT) {
    // ---- stage K (row-major, vectorized: 8x uint4 per thread) ----
    const __bf16* ksrc =
        Kb + (size_t)(b * S_SZ + kt0 + kr) * (H_SZ * DHEAD) + h * DHEAD + kc;
#pragma unroll
    for (int i = 0; i < 8; i++)
      *reinterpret_cast<uint4*>(&Kt[kr][kc + i * 8]) =
          *reinterpret_cast<const uint4*>(ksrc + i * 8);
    // ---- stage V transposed: 2 k-rows x 32 d, pack (k,k+1) dwords ----
    {
      const __bf16* v0 =
          Vb + (size_t)(b * S_SZ + kt0 + 2 * vkp) * (H_SZ * DHEAD) + h * DHEAD + vdg * 32;
#pragma unroll
      for (int g = 0; g < 4; g++) {   // 8 d at a time
        const uint4 ra = *reinterpret_cast<const uint4*>(v0 + g * 8);
        const uint4 rc = *reinterpret_cast<const uint4*>(v0 + H_SZ * DHEAD + g * 8);
        const unsigned short* pa = reinterpret_cast<const unsigned short*>(&ra);
        const unsigned short* pc = reinterpret_cast<const unsigned short*>(&rc);
#pragma unroll
        for (int j = 0; j < 8; j++) {
          *reinterpret_cast<unsigned int*>(&VtT[vdg * 32 + g * 8 + j][2 * vkp]) =
              (unsigned int)pa[j] | ((unsigned int)pc[j] << 16);
        }
      }
    }
    __syncthreads();

    // scores: four 16x16 tiles (key groups j*16)
    f32x8 s[4];
#pragma unroll
    for (int j = 0; j < 4; j++) s[j] = zero;
#pragma unroll
    for (int dc = 0; dc < 4; dc++) {
#pragma unroll
      for (int j = 0; j < 4; j++) {
        bf16x16 kf;
#pragma unroll
        for (int e = 0; e < 16; e++)
          kf[e] = Kt[j * 16 + l15][dc * 32 + half * 16 + e];   // contiguous 32B
        s[j] = wmma_bf16(qf[dc], kf, s[j]);
      }
    }

    // online softmax per query row (row = 8*half + r, spread across 16 lanes)
#pragma unroll
    for (int r = 0; r < 8; r++) {
      const float a0 = s[0][r] * scale;
      const float a1 = s[1][r] * scale;
      const float a2 = s[2][r] * scale;
      const float a3 = s[3][r] * scale;
      float mt = fmaxf(fmaxf(a0, a1), fmaxf(a2, a3));
#pragma unroll
      for (int off = 1; off < 16; off <<= 1)
        mt = fmaxf(mt, __shfl_xor(mt, off, 32));
      const float mnew  = fmaxf(mrow[r], mt);
      const float alpha = __expf(mrow[r] - mnew);
      const float p0 = __expf(a0 - mnew);
      const float p1 = __expf(a1 - mnew);
      const float p2 = __expf(a2 - mnew);
      const float p3 = __expf(a3 - mnew);
      float ps = (p0 + p1) + (p2 + p3);
#pragma unroll
      for (int off = 1; off < 16; off <<= 1)
        ps += __shfl_xor(ps, off, 32);
      lrow[r] = lrow[r] * alpha + ps;
      mrow[r] = mnew;
#pragma unroll
      for (int db = 0; db < 8; db++) oacc[db][r] *= alpha;
      Pt[wid][half * 8 + r][l15]      = (__bf16)p0;
      Pt[wid][half * 8 + r][16 + l15] = (__bf16)p1;
      Pt[wid][half * 8 + r][32 + l15] = (__bf16)p2;
      Pt[wid][half * 8 + r][48 + l15] = (__bf16)p3;
    }

    // P (16x64) @ V (64x128): P transposed through wave-private LDS tile
    bf16x16 pf[2];
#pragma unroll
    for (int c = 0; c < 2; c++) {
#pragma unroll
      for (int e = 0; e < 16; e++) {
        const int kk = half * 16 + e;
        pf[c][e] = Pt[wid][l15][c * 32 + kk];
      }
    }
#pragma unroll
    for (int db = 0; db < 8; db++) {
#pragma unroll
      for (int c = 0; c < 2; c++) {
        bf16x16 vf;
#pragma unroll
        for (int e = 0; e < 16; e++)
          vf[e] = VtT[db * 16 + l15][c * 32 + half * 16 + e];  // contiguous 32B
        oacc[db] = wmma_bf16(pf[c], vf, oacc[db]);
      }
    }
    __syncthreads();
  }

  // normalize (one reciprocal per row, then multiplies) and write out
  float invl[8];
#pragma unroll
  for (int r = 0; r < 8; r++) invl[r] = 1.0f / lrow[r];
#pragma unroll
  for (int db = 0; db < 8; db++) {
#pragma unroll
    for (int r = 0; r < 8; r++) {
      const float v = oacc[db][r] * invl[r];
      const int row = b * S_SZ + q0 + half * 8 + r;
      const int col = h * DHEAD + db * 16 + l15;
      O[(size_t)row * (H_SZ * DHEAD) + col] = (__bf16)v;
    }
  }
}

// ---------------------------------------------------------------------------
// Host launcher
// ---------------------------------------------------------------------------
extern "C" void kernel_launch(void* const* d_in, const int* in_sizes, int n_in,
                              void* d_out, int out_size, void* d_ws, size_t ws_size,
                              hipStream_t stream) {
  (void)in_sizes; (void)n_in; (void)out_size; (void)ws_size;

  const float* x     = (const float*)d_in[0];
  const float* W_dkv = (const float*)d_in[1];
  const float* b_dkv = (const float*)d_in[2];
  const float* W_dq  = (const float*)d_in[3];
  const float* b_dq  = (const float*)d_in[4];
  const float* W_uk  = (const float*)d_in[5];
  const float* b_uk  = (const float*)d_in[6];
  const float* W_uv  = (const float*)d_in[7];
  const float* b_uv  = (const float*)d_in[8];
  const float* W_uq  = (const float*)d_in[9];
  const float* b_uq  = (const float*)d_in[10];
  const float* W_qr  = (const float*)d_in[11];
  const float* b_qr  = (const float*)d_in[12];
  const float* W_kr  = (const float*)d_in[13];
  const float* b_kr  = (const float*)d_in[14];
  const float* W_o   = (const float*)d_in[15];
  const float* b_o   = (const float*)d_in[16];

  float* out_main = (float*)d_out;                           // [4096, 2048]
  float* out_ckv  = out_main + (size_t)MROWS * DIN;          // [4096, 512]
  float* out_krot = out_ckv + (size_t)MROWS * DKV;           // [4096, 16, 32]

  // workspace allocator (256B aligned)
  char* wsp = (char*)d_ws;
  size_t off = 0;
  auto alloc = [&](size_t bytes) -> void* {
    void* p = wsp + off;
    off += (bytes + 255) & ~(size_t)255;
    return p;
  };

  __bf16* x_bf     = (__bf16*)alloc((size_t)MROWS * DIN * 2);
  __bf16* wdkv_bf  = (__bf16*)alloc((size_t)DIN * DKV * 2);
  __bf16* wdq_bf   = (__bf16*)alloc((size_t)DIN * DKV * 2);
  __bf16* wuk_bf   = (__bf16*)alloc((size_t)DKV * (H_SZ * 96) * 2);
  __bf16* wuv_bf   = (__bf16*)alloc((size_t)DKV * (H_SZ * DHEAD) * 2);
  __bf16* wuq_bf   = (__bf16*)alloc((size_t)DKV * (H_SZ * 96) * 2);
  __bf16* wqr_bf   = (__bf16*)alloc((size_t)DKV * (H_SZ * 32) * 2);
  __bf16* wkr_bf   = (__bf16*)alloc((size_t)DIN * (H_SZ * 32) * 2);
  __bf16* wo_bf    = (__bf16*)alloc((size_t)(H_SZ * DHEAD) * DIN * 2);
  __bf16* ckv_bf   = (__bf16*)alloc((size_t)MROWS * DKV * 2);
  __bf16* cq_bf    = (__bf16*)alloc((size_t)MROWS * DKV * 2);
  __bf16* q_buf    = (__bf16*)alloc((size_t)MROWS * (H_SZ * DHEAD) * 2);
  __bf16* k_buf    = (__bf16*)alloc((size_t)MROWS * (H_SZ * DHEAD) * 2);
  __bf16* v_buf    = (__bf16*)alloc((size_t)MROWS * (H_SZ * DHEAD) * 2);
  __bf16* attn_bf  = (__bf16*)alloc((size_t)MROWS * (H_SZ * DHEAD) * 2);
  float*  qrot_pre = (float*)alloc((size_t)MROWS * (H_SZ * 32) * 4);
  float*  krot_pre = (float*)alloc((size_t)MROWS * (H_SZ * 32) * 4);

  auto conv = [&](const float* src, __bf16* dst, size_t n) {
    f32_to_bf16_kernel<<<dim3((unsigned)((n + 255) / 256)), dim3(256), 0, stream>>>(
        src, dst, (int)n);
  };
  conv(x,     x_bf,    (size_t)MROWS * DIN);
  conv(W_dkv, wdkv_bf, (size_t)DIN * DKV);
  conv(W_dq,  wdq_bf,  (size_t)DIN * DKV);
  conv(W_uk,  wuk_bf,  (size_t)DKV * (H_SZ * 96));
  conv(W_uv,  wuv_bf,  (size_t)DKV * (H_SZ * DHEAD));
  conv(W_uq,  wuq_bf,  (size_t)DKV * (H_SZ * 96));
  conv(W_qr,  wqr_bf,  (size_t)DKV * (H_SZ * 32));
  conv(W_kr,  wkr_bf,  (size_t)DIN * (H_SZ * 32));
  conv(W_o,   wo_bf,   (size_t)(H_SZ * DHEAD) * DIN);

  const dim3 blk(256);
  auto grid = [](int n) { return dim3((unsigned)(n / 128), (unsigned)(MROWS / 64)); };

  // c_kv = x @ W_dkv + b_dkv  (fp32 to d_out, bf16 for downstream GEMMs)
  gemm_bf16_kernel<512, 2048, true, true, 512, 512, 512, 0>
      <<<grid(512), blk, 0, stream>>>(x_bf, wdkv_bf, b_dkv, out_ckv, ckv_bf);
  // c_q = x @ W_dq + b_dq
  gemm_bf16_kernel<512, 2048, false, true, 512, 512, 512, 0>
      <<<grid(512), blk, 0, stream>>>(x_bf, wdq_bf, b_dq, nullptr, cq_bf);
  // k_nope -> k_buf[:, h*128 + 0..95]
  gemm_bf16_kernel<1536, 512, false, true, 2048, 96, 128, 0>
      <<<grid(1536), blk, 0, stream>>>(ckv_bf, wuk_bf, b_uk, nullptr, k_buf);
  // v -> v_buf (plain)
  gemm_bf16_kernel<2048, 512, false, true, 2048, 2048, 2048, 0>
      <<<grid(2048), blk, 0, stream>>>(ckv_bf, wuv_bf, b_uv, nullptr, v_buf);
  // q_base -> q_buf[:, h*128 + 0..95]
  gemm_bf16_kernel<1536, 512, false, true, 2048, 96, 128, 0>
      <<<grid(1536), blk, 0, stream>>>(cq_bf, wuq_bf, b_uq, nullptr, q_buf);
  // q_rot_pre (fp32) = c_q @ W_qr + b_qr
  gemm_bf16_kernel<512, 512, true, false, 512, 512, 512, 0>
      <<<grid(512), blk, 0, stream>>>(cq_bf, wqr_bf, b_qr, qrot_pre, nullptr);
  // k_rot_pre (fp32) = x @ W_kr + b_kr
  gemm_bf16_kernel<512, 2048, true, false, 512, 512, 512, 0>
      <<<grid(512), blk, 0, stream>>>(x_bf, wkr_bf, b_kr, krot_pre, nullptr);

  // rotary: fills q_buf/k_buf cols 96..127 and fp32 k_rot output
  {
    const int total = MROWS * H_SZ * 16;
    rope_kernel<<<dim3((unsigned)((total + 255) / 256)), dim3(256), 0, stream>>>(
        qrot_pre, krot_pre, q_buf, k_buf, out_krot);
  }

  // attention
  mla_attention_kernel<<<dim3(S_SZ / 64, H_SZ, B_SZ), dim3(128), 0, stream>>>(
      q_buf, k_buf, v_buf, attn_bf);

  // out = attn @ W_o + b_o (fp32 to d_out)
  gemm_bf16_kernel<2048, 2048, true, false, 2048, 2048, 2048, 0>
      <<<grid(2048), blk, 0, stream>>>(attn_bf, wo_bf, b_o, out_main, nullptr);
}
